// CrossAttention_49005576847528
// MI455X (gfx1250) — matrix operand order repaired and checked by reading, not compiled, not run on hardware
//
#include <hip/hip_runtime.h>

// ---------------------------------------------------------------------------
// CrossAttention for MI455X (gfx1250, wave32, WMMA f16 -> f32 accum)
// B=2, S1=S2=2048, D1=D2=1024, H=16, K=V=64
// Round 2: attention K/V/weight tiles staged via Tensor Data Mover
//          (tensor_load_to_lds + s_wait_tensorcnt, double buffered).
// ---------------------------------------------------------------------------

typedef _Float16 v16h __attribute__((ext_vector_type(16)));
typedef _Float16 v4h  __attribute__((ext_vector_type(4)));
typedef float    v8f  __attribute__((ext_vector_type(8)));
typedef unsigned int u32x4 __attribute__((ext_vector_type(4)));
typedef int          i32x4 __attribute__((ext_vector_type(4)));
typedef int          i32x8 __attribute__((ext_vector_type(8)));

#define B_   2
#define S1_  2048
#define S2_  2048
#define D1_  1024
#define D2_  1024
#define H_   16
#define KD_  64
#define VD_  64
#define HK_  1024   // H_*KD_
#define HV_  1024   // H_*VD_

#if defined(__HIP_DEVICE_COMPILE__) && __has_builtin(__builtin_amdgcn_tensor_load_to_lds)
#define HAVE_TDM 1
#else
#define HAVE_TDM 0
#endif

// --------------------------------------------------------------------------
// fp32 -> f16 conversion (vectorized x4)
// --------------------------------------------------------------------------
__global__ __launch_bounds__(256) void cvt_f32_f16(const float* __restrict__ src,
                                                   _Float16* __restrict__ dst,
                                                   int n4) {
  int i = blockIdx.x * blockDim.x + threadIdx.x;
  if (i < n4) {
    float4 f = ((const float4*)src)[i];
    v4h h;
    h[0] = (_Float16)f.x; h[1] = (_Float16)f.y;
    h[2] = (_Float16)f.z; h[3] = (_Float16)f.w;
    ((v4h*)dst)[i] = h;
  }
}

// --------------------------------------------------------------------------
// TDM helper: 2-D tile DMA  global -> LDS  (D# per CDNA5 ISA)
//   data_size_log: 0=1B, 1=2B, 2=4B
//   tile_d0 elements contiguous per row, tile_d1 rows, row stride in elements
// --------------------------------------------------------------------------
#if HAVE_TDM
__device__ __forceinline__ void tdm_load_2d(unsigned lds_addr, const void* gptr,
                                            unsigned data_size_log,
                                            unsigned tile_d0, unsigned tile_d1,
                                            unsigned long long stride_elems) {
  unsigned long long ga = (unsigned long long)gptr;
  u32x4 g0;
  g0[0] = 1u;                                              // count=1, user desc
  g0[1] = lds_addr;                                        // LDS byte address
  g0[2] = (unsigned)(ga & 0xFFFFFFFFu);                    // global_addr[31:0]
  g0[3] = (unsigned)((ga >> 32) & 0x01FFFFFFu) | (2u << 30); // addr[56:32], type=2
  i32x8 g1;
  g1[0] = (int)(data_size_log << 16);                      // wg_mask=0, data_size
  g1[1] = (int)((tile_d0 & 0xFFFFu) << 16);                // tensor_dim0[15:0]
  g1[2] = (int)(((tile_d0 >> 16) & 0xFFFFu) |
                ((tile_d1 & 0xFFFFu) << 16));              // tdim0 hi / tdim1 lo
  g1[3] = (int)(((tile_d1 >> 16) & 0xFFFFu) |
                ((tile_d0 & 0xFFFFu) << 16));              // tdim1 hi / tile_dim0
  g1[4] = (int)(tile_d1 & 0xFFFFu);                        // tile_dim1, tile_dim2=0
  g1[5] = (int)(stride_elems & 0xFFFFFFFFu);               // dim0_stride[31:0]
  g1[6] = (int)((stride_elems >> 32) & 0xFFFFu);           // dim0_stride[47:32]
  g1[7] = 0;
  i32x4 z4 = {0, 0, 0, 0};
#if defined(__clang_major__) && (__clang_major__ >= 23)
  i32x8 z8 = {0, 0, 0, 0, 0, 0, 0, 0};
  __builtin_amdgcn_tensor_load_to_lds(g0, g1, z4, z4, z8, 0);
#else
  __builtin_amdgcn_tensor_load_to_lds(g0, g1, z4, z4, 0);
#endif
}
#endif

__device__ __forceinline__ unsigned lds_off(const void* p) {
  // flat->LDS aperture truncates to addr[31:0]; compile-time plumbing for D#
  return (unsigned)(unsigned long long)p;
}

// --------------------------------------------------------------------------
// Tiled WMMA GEMM:  C[M,N] = A[M,Kd] (f16, row-major) @ W[N,Kd]^T (f16)
// Block: 256 thr = 8 waves, block tile 64(M) x 128(N); wave tile 16 x 64.
// Output: f16 (C16) or f32 (+bias) (C32).
// --------------------------------------------------------------------------
__global__ __launch_bounds__(256) void gemm_wmma_f16(const _Float16* __restrict__ A,
                                                     const _Float16* __restrict__ W,
                                                     _Float16* __restrict__ C16,
                                                     float* __restrict__ C32,
                                                     const float* __restrict__ bias,
                                                     int M, int N, int Kd) {
  __shared__ _Float16 At[64][40];   // 80B row stride -> 16B aligned segments
  __shared__ _Float16 Bt[128][40];

  const int tid  = threadIdx.x;
  const int lane = tid & 31;
  const int wave = tid >> 5;
  const int wy   = wave >> 1;   // 0..3  (M strip)
  const int wx   = wave & 1;    // 0..1  (N strip)
  const int m0   = blockIdx.y * 64;
  const int n0   = blockIdx.x * 128;

  const int hi   = lane >> 4;     // half-wave
  const int l15  = lane & 15;
  const int khA  = hi * 8;        // A-fragment k half
  const int kbB  = hi * 16;       // unused; B fragment indexes with khA

  v8f acc[4] = {};

  for (int kk = 0; kk < Kd; kk += 32) {
    // ---- stage A tile 64x32 (each thread: 16B) ----
    {
      int row = tid >> 2;
      int cb  = (tid & 3) * 8;
      const uint4* src = (const uint4*)(A + (size_t)(m0 + row) * Kd + kk + cb);
      __builtin_prefetch(A + (size_t)(m0 + row) * Kd + kk + 32 + cb, 0, 1);
      *(uint4*)&At[row][cb] = src[0];
    }
    // ---- stage B tile 128x32 (each thread: 32B) ----
    {
      int nr = tid >> 1;
      int cb = (tid & 1) * 16;
      const uint4* src = (const uint4*)(W + (size_t)(n0 + nr) * Kd + kk + cb);
      __builtin_prefetch(W + (size_t)(n0 + nr) * Kd + kk + 32 + cb, 0, 1);
      *(uint4*)&Bt[nr][cb]     = src[0];
      *(uint4*)&Bt[nr][cb + 8] = src[1];
    }
    __syncthreads();

    // ---- A fragment (16x32, ISA layout) ----
    v16h a;
    {
      const _Float16* ar = &At[wy * 16 + l15][0];
      #pragma unroll
      for (int i = 0; i < 8; ++i)  a[i] = ar[khA + i];
      #pragma unroll
      for (int i = 8; i < 16; ++i) a[i] = ar[16 + khA + (i - 8)];
    }
    // ---- 4 B fragments + WMMA ----
    #pragma unroll
    for (int nt = 0; nt < 4; ++nt) {
      v16h b;
      const _Float16* br = &Bt[wx * 64 + nt * 16 + l15][0];
      #pragma unroll
      for (int i = 0; i < 16; ++i) b[i] = br[(i < 8) ? (khA + i) : (16 + khA + (i - 8))];
      acc[nt] = __builtin_amdgcn_wmma_f32_16x16x32_f16(
          false, a, false, b, (short)0, acc[nt], false, false);
    }
    __syncthreads();
  }

  // ---- epilogue ----
  #pragma unroll
  for (int nt = 0; nt < 4; ++nt) {
    #pragma unroll
    for (int r = 0; r < 8; ++r) {
      int m = m0 + wy * 16 + r + hi * 8;
      int n = n0 + wx * 64 + nt * 16 + l15;
      float v = acc[nt][r];
      if (C32) {
        C32[(size_t)m * N + n] = v + (bias ? bias[n] : 0.0f);
      } else {
        C16[(size_t)m * N + n] = (_Float16)v;
      }
    }
  }
}

// --------------------------------------------------------------------------
// Flash-style attention, one wave per (b, h, 16-query tile).
//  S = (Q K^T) * 1/8, mask(-inf), * weight_matrix, online softmax, O = P V.
// K/V/weight tiles staged by the Tensor Data Mover into double-buffered LDS;
// the single wave gives clean per-wave TENSORcnt pipelining:
//   issue(chunk i+1) -> s_wait_tensorcnt(3) -> compute(chunk i)
// --------------------------------------------------------------------------
__global__ __launch_bounds__(32) void attn_wmma(const _Float16* __restrict__ Q,
                                                const _Float16* __restrict__ Kh,
                                                const _Float16* __restrict__ Vh,
                                                const float* __restrict__ WM,
                                                const int* __restrict__ mask,
                                                _Float16* __restrict__ O) {
  __shared__ _Float16 Kt[2][32][64];   // [buf][key][head-dim]
  __shared__ _Float16 Vt[2][32][64];   // [buf][key][v-dim]
  __shared__ float    Wt[2][16][32];   // [buf][query][key]
  __shared__ _Float16 Pld[16][40];

  const int lane = threadIdx.x;
  const int hi   = lane >> 4;
  const int l15  = lane & 15;
  const int q0   = blockIdx.x * 16;
  const int h    = blockIdx.y;
  const int b    = blockIdx.z;
  const float scale = 0.125f;           // 1/sqrt(64)
  const float NEG_INF = -__builtin_inff();
  const int NCH = S2_ / 32;

  const _Float16* Kbase = Kh + (size_t)(b * S2_) * HK_ + h * 64;
  const _Float16* Vbase = Vh + (size_t)(b * S2_) * HV_ + h * 64;
  const size_t wbase = ((size_t)(b * H_ + h) * S1_ + q0) * S2_;

  // ---- Q fragments (16 x 64 = two 16x32 A fragments), held in registers ----
  v16h aq0, aq1;
  {
    const _Float16* qr = Q + ((size_t)(b * S1_) + q0 + l15) * HK_ + h * 64;
    int kh = hi * 8;
    #pragma unroll
    for (int i = 0; i < 8; ++i)  { aq0[i] = qr[kh + i];            aq1[i] = qr[32 + kh + i]; }
    #pragma unroll
    for (int i = 8; i < 16; ++i) { aq0[i] = qr[16 + kh + (i - 8)]; aq1[i] = qr[48 + kh + (i - 8)]; }
  }

  // ---- query mask (rows handled by this lane: r + hi*8) ----
  bool mz[8];
  #pragma unroll
  for (int r = 0; r < 8; ++r)
    mz[r] = (mask[b * S1_ + q0 + r + hi * 8] == 0);

  v8f o0 = {}, o1 = {}, o2 = {}, o3 = {};
  float m_run[8], l_run[8];
  #pragma unroll
  for (int r = 0; r < 8; ++r) { m_run[r] = NEG_INF; l_run[r] = 0.0f; }

  const int kbB = hi * 16;

  // ---- tile stage helper: 3 TDM 2-D DMAs per 32-key chunk ----
  auto stage = [&](int s2, int buf) {
#if HAVE_TDM
    tdm_load_2d(lds_off(&Kt[buf][0][0]), Kbase + (size_t)s2 * HK_, 1, 64, 32, HK_);
    tdm_load_2d(lds_off(&Vt[buf][0][0]), Vbase + (size_t)s2 * HV_, 1, 64, 32, HV_);
    tdm_load_2d(lds_off(&Wt[buf][0][0]), WM + wbase + s2,          2, 32, 16, S2_);
#else
    // fallback: cooperative per-lane copies
    {
      const uint4* ks = (const uint4*)(Kbase + (size_t)(s2 + lane) * HK_);
      uint4* kd = (uint4*)&Kt[buf][lane][0];
      const uint4* vs = (const uint4*)(Vbase + (size_t)(s2 + lane) * HV_);
      uint4* vd = (uint4*)&Vt[buf][lane][0];
      #pragma unroll
      for (int i = 0; i < 8; ++i) { kd[i] = ks[i]; vd[i] = vs[i]; }
    }
    {
      const float2* ws = (const float2*)(WM + wbase + (size_t)l15 * S2_ + s2 + hi * 16);
      float2* wd = (float2*)&Wt[buf][l15][hi * 16];
      #pragma unroll
      for (int i = 0; i < 8; ++i) wd[i] = ws[i];
    }
#endif
  };

  stage(0, 0);

  for (int ic = 0; ic < NCH; ++ic) {
    const int cur = ic & 1;
    const int s2  = ic * 32;

    if (ic + 1 < NCH) {
      stage(s2 + 32, cur ^ 1);
#if HAVE_TDM
      __builtin_amdgcn_s_wait_tensorcnt(3);   // chunk ic's 3 DMAs complete
#endif
    } else {
#if HAVE_TDM
      __builtin_amdgcn_s_wait_tensorcnt(0);
#endif
    }
    __syncthreads();   // order TDM/LDS writes vs the ds reads below

    // ---- scores: two 16x16 tiles over 32 keys, K from LDS ----
    v8f sc0 = {}, sc1 = {};
    {
      const _Float16* k0 = &Kt[cur][l15][0];
      const _Float16* k1 = &Kt[cur][16 + l15][0];
      v16h b00, b01, b10, b11;
      #pragma unroll
      for (int i = 0; i < 16; ++i) {
        int kx = (i < 8) ? (hi * 8 + i) : (16 + hi * 8 + (i - 8));
        b00[i] = k0[kx];       b01[i] = k0[32 + kx];
        b10[i] = k1[kx];       b11[i] = k1[32 + kx];
      }
      sc0 = __builtin_amdgcn_wmma_f32_16x16x32_f16(false, aq0, false, b00, (short)0, sc0, false, false);
      sc0 = __builtin_amdgcn_wmma_f32_16x16x32_f16(false, aq1, false, b01, (short)0, sc0, false, false);
      sc1 = __builtin_amdgcn_wmma_f32_16x16x32_f16(false, aq0, false, b10, (short)0, sc1, false, false);
      sc1 = __builtin_amdgcn_wmma_f32_16x16x32_f16(false, aq1, false, b11, (short)0, sc1, false, false);
    }

    // ---- scale, mask, multiplicative bias (from LDS tile), online softmax ----
    float p0[8], p1[8];
    #pragma unroll
    for (int r = 0; r < 8; ++r) {
      int m = r + hi * 8;
      float s0 = sc0[r] * scale;
      float s1 = sc1[r] * scale;
      if (mz[r]) { s0 = NEG_INF; s1 = NEG_INF; }
      s0 *= Wt[cur][m][l15];
      s1 *= Wt[cur][m][16 + l15];

      float t = fmaxf(s0, s1);
      t = fmaxf(t, __shfl_xor(t, 1, 32));
      t = fmaxf(t, __shfl_xor(t, 2, 32));
      t = fmaxf(t, __shfl_xor(t, 4, 32));
      t = fmaxf(t, __shfl_xor(t, 8, 32));
      float nm = fmaxf(m_run[r], t);
      float f  = __expf(m_run[r] - nm);
      p0[r] = __expf(s0 - nm);
      p1[r] = __expf(s1 - nm);
      float rs = p0[r] + p1[r];
      rs += __shfl_xor(rs, 1, 32);
      rs += __shfl_xor(rs, 2, 32);
      rs += __shfl_xor(rs, 4, 32);
      rs += __shfl_xor(rs, 8, 32);
      l_run[r] = l_run[r] * f + rs;
      m_run[r] = nm;
      o0[r] *= f; o1[r] *= f; o2[r] *= f; o3[r] *= f;
    }

    // ---- P: C-layout -> A-fragment via LDS ----
    #pragma unroll
    for (int r = 0; r < 8; ++r) {
      int m = r + hi * 8;
      Pld[m][l15]      = (_Float16)p0[r];
      Pld[m][16 + l15] = (_Float16)p1[r];
    }
    __syncthreads();
    v16h ap;
    {
      const _Float16* pr = &Pld[l15][0];
      int kh = hi * 8;
      #pragma unroll
      for (int i = 0; i < 8; ++i)  ap[i] = pr[kh + i];
      #pragma unroll
      for (int i = 8; i < 16; ++i) ap[i] = pr[16 + kh + (i - 8)];
    }
    __syncthreads();

    // ---- PV: 4 WMMAs, V from the staged LDS tile ----
    #pragma unroll
    for (int nt = 0; nt < 4; ++nt) {
      v16h bv;
      #pragma unroll
      for (int i = 0; i < 16; ++i)
        bv[i] = Vt[cur][(i < 8) ? (hi * 8 + i) : (16 + hi * 8 + (i - 8))][nt * 16 + l15];
      v8f* op = (nt == 0) ? &o0 : (nt == 1) ? &o1 : (nt == 2) ? &o2 : &o3;
      *op = __builtin_amdgcn_wmma_f32_16x16x32_f16(false, ap, false, bv, (short)0, *op, false, false);
    }
  }

  // ---- normalize + store (head-major concat layout [B*S1, H*64]) ----
  #pragma unroll
  for (int r = 0; r < 8; ++r) {
    int m = q0 + r + hi * 8;
    float inv = 1.0f / l_run[r];
    size_t base = ((size_t)(b * S1_) + m) * HV_ + h * 64 + l15;
    O[base]      = (_Float16)(o0[r] * inv);
    O[base + 16] = (_Float16)(o1[r] * inv);
    O[base + 32] = (_Float16)(o2[r] * inv);
    O[base + 48] = (_Float16)(o3[r] * inv);
  }
}

// --------------------------------------------------------------------------
// Host-side orchestration
// --------------------------------------------------------------------------
extern "C" void kernel_launch(void* const* d_in, const int* in_sizes, int n_in,
                              void* d_out, int out_size, void* d_ws, size_t ws_size,
                              hipStream_t stream) {
  (void)in_sizes; (void)n_in; (void)out_size; (void)ws_size;

  const float* x1 = (const float*)d_in[0];
  const float* x2 = (const float*)d_in[1];
  const float* wm = (const float*)d_in[2];
  const int*   mk = (const int*)  d_in[3];
  const float* Wq = (const float*)d_in[4];
  const float* Wk = (const float*)d_in[5];
  const float* Wv = (const float*)d_in[6];
  const float* Wo = (const float*)d_in[7];
  const float* bo = (const float*)d_in[8];
  float* out = (float*)d_out;

  char* wsp = (char*)d_ws;
  auto take = [&](size_t bytes) {
    char* p = wsp;
    wsp += (bytes + 255) & ~(size_t)255;
    return p;
  };
  const size_t NX  = (size_t)B_ * S1_ * D1_;   // 4M elems
  const size_t NW  = (size_t)HK_ * D1_;        // 1M elems
  _Float16* x1h = (_Float16*)take(NX * 2);
  _Float16* x2h = (_Float16*)take(NX * 2);
  _Float16* Wqh = (_Float16*)take(NW * 2);
  _Float16* Wkh = (_Float16*)take(NW * 2);
  _Float16* Wvh = (_Float16*)take(NW * 2);
  _Float16* Woh = (_Float16*)take(NW * 2);
  _Float16* Qh  = (_Float16*)take((size_t)B_ * S1_ * HK_ * 2);
  _Float16* Kc  = (_Float16*)take((size_t)B_ * S2_ * HK_ * 2);
  _Float16* Vc  = (_Float16*)take((size_t)B_ * S2_ * HV_ * 2);
  _Float16* Oh  = (_Float16*)take((size_t)B_ * S1_ * HV_ * 2);

  // 1) fp32 -> f16 conversions
  auto cvt = [&](const float* s, _Float16* d, size_t n) {
    int n4 = (int)(n / 4);
    cvt_f32_f16<<<(n4 + 255) / 256, 256, 0, stream>>>(s, d, n4);
  };
  cvt(x1, x1h, NX);
  cvt(x2, x2h, NX);
  cvt(Wq, Wqh, NW);
  cvt(Wk, Wkh, NW);
  cvt(Wv, Wvh, NW);
  cvt(Wo, Woh, NW);

  // 2) QKV projections: [4096,1024] = [4096,1024] @ [1024,1024]^T
  const int M = B_ * S1_, N = HK_, Kd = D1_;
  dim3 ggrid(N / 128, M / 64);
  gemm_wmma_f16<<<ggrid, 256, 0, stream>>>(x1h, Wqh, Qh, nullptr, nullptr, M, N, Kd);
  gemm_wmma_f16<<<ggrid, 256, 0, stream>>>(x2h, Wkh, Kc, nullptr, nullptr, M, N, Kd);
  gemm_wmma_f16<<<ggrid, 256, 0, stream>>>(x2h, Wvh, Vc, nullptr, nullptr, M, N, Kd);

  // 3) attention (flash-style; TDM streams K/V/weight tiles into LDS)
  dim3 agrid(S1_ / 16, H_, B_);
  attn_wmma<<<agrid, 32, 0, stream>>>(Qh, Kc, Vc, wm, mk, Oh);

  // 4) output projection with bias: out[4096,1024] = Oh @ Wo^T + bo
  gemm_wmma_f16<<<ggrid, 256, 0, stream>>>(Oh, Woh, nullptr, out, bo, M, HV_, D1_);
}
